// SelfCorrelation4_11613591569062
// MI455X (gfx1250) — matrix operand order repaired and checked by reading, not compiled, not run on hardware
//
#include <hip/hip_runtime.h>

typedef unsigned int   u32;
typedef unsigned short u16;
typedef __attribute__((ext_vector_type(16))) __bf16 v16bf;
typedef __attribute__((ext_vector_type(8)))  float  v8f;

#define NB      2
#define C_IN    256
#define CQ      64
#define NPOS    6272      // 8*28*28, = 392*16
#define NTILES  392
#define NCHUNKS 196       // NPOS/32
#define WPB     8         // waves per block (one WGP), 392 = 49*8
#define LOG2E   1.4426950408889634f
#define LSE_SHIFT_L2 (20.0f * LOG2E)  // logits ~ N(0,8): exp(s-20) is safe in f32

struct alignas(16) U4 { u32 x, y, z, w; };
union BF16Tile { v16bf v; U4 q[2]; u32 u[8]; };

__device__ __forceinline__ U4 ld16(const u16* p) { return *(const U4*)p; }

__device__ __forceinline__ u16 f2bf(float f) {
  u32 u = __builtin_bit_cast(u32, f);
  u += 0x7FFFu + ((u >> 16) & 1u);   // round-to-nearest-even
  return (u16)(u >> 16);
}

// raw v_exp_f32 / v_log_f32 (no libm denormal guard; flush-to-zero is the
// correct behavior for vanishing attention weights)
__device__ __forceinline__ float exp2_raw(float x) { return __builtin_amdgcn_exp2f(x); }
__device__ __forceinline__ float log2_raw(float x) { return __builtin_amdgcn_logf(x); }

__device__ __forceinline__ v8f wmma_bf16(const BF16Tile& a, const BF16Tile& b, v8f c) {
  return __builtin_amdgcn_wmma_f32_16x16x32_bf16(false, a.v, false, b.v,
                                                 (short)0, c, false, false);
}

// ---------------- Kernel 1: QKV projections (f32 math, bf16 out) ----------------
// QbT/KbT: [B][N][64] (k-contiguous, WMMA-operand friendly), Vb: [B][C][N]
__global__ void proj_kernel(const float* __restrict__ x,
                            const float* __restrict__ W1, const float* __restrict__ b1,
                            const float* __restrict__ W2, const float* __restrict__ b2,
                            const float* __restrict__ W3, const float* __restrict__ b3,
                            u16* __restrict__ QbT, u16* __restrict__ KbT,
                            u16* __restrict__ Vb) {
  const int n = blockIdx.x * blockDim.x + threadIdx.x;
  if (n >= NPOS) return;
  const int d = blockIdx.y;          // 0..383 : 0-63 Q, 64-127 K, 128-383 V
  const int b = blockIdx.z;
  const float* W; float bias; int row;
  if (d < CQ)        { W = W1; bias = b1[d];        row = d; }
  else if (d < 2*CQ) { W = W2; bias = b2[d - CQ];   row = d - CQ; }
  else               { W = W3; bias = b3[d - 2*CQ]; row = d - 2*CQ; }
  const float* __restrict__ xc = x + (size_t)b * C_IN * NPOS + n;
  const float* __restrict__ wr = W + (size_t)row * C_IN;
  float acc = bias;
  #pragma unroll 8
  for (int c = 0; c < C_IN; ++c) acc = fmaf(wr[c], xc[(size_t)c * NPOS], acc);
  const u16 bf = f2bf(acc);
  if (d < CQ)        QbT[((size_t)b * NPOS + n) * CQ + d] = bf;
  else if (d < 2*CQ) KbT[((size_t)b * NPOS + n) * CQ + (d - CQ)] = bf;
  else               Vb[((size_t)b * C_IN + (d - 2*CQ)) * NPOS + n] = bf;
}

// ---------------- Kernel 2: per-row (n) log2-sum-exp2 of S = K^T Q ----------------
// one wave per 16-row n-tile; streams all 392 m-tiles via WMMA.
// Stores off2[n] = -(SHIFT_L2 + log2(sum exp2(s*log2e - SHIFT_L2)))  (= -logZ*log2e)
__global__ void __launch_bounds__(32 * WPB)
rowstats_kernel(const u16* __restrict__ QbT, const u16* __restrict__ KbT,
                float* __restrict__ off2) {
  const int lane = threadIdx.x & 31;
  const int wv   = threadIdx.x >> 5;
  const int ntile = blockIdx.x * WPB + wv;
  const int b  = blockIdx.y;
  const int lh = lane >> 4;       // half-wave select
  const int lm = lane & 15;

  // A operand: K^T tile (rows n, k=0..63), fixed for this wave
  const u16* Krow = KbT + ((size_t)b * NPOS + (ntile * 16 + lm)) * CQ + 8 * lh;
  BF16Tile a0, a1;
  a0.q[0] = ld16(Krow);      a0.q[1] = ld16(Krow + 16);
  a1.q[0] = ld16(Krow + 32); a1.q[1] = ld16(Krow + 48);

  float sm[8];
  #pragma unroll
  for (int r = 0; r < 8; ++r) sm[r] = 0.0f;

  const u16* Qbase = QbT + (size_t)b * NPOS * CQ + 8 * lh;
  for (int mt = 0; mt < NTILES; ++mt) {
    const u16* Qrow = Qbase + (size_t)(mt * 16 + lm) * CQ;
    BF16Tile q0, q1;
    q0.q[0] = ld16(Qrow);      q0.q[1] = ld16(Qrow + 16);
    q1.q[0] = ld16(Qrow + 32); q1.q[1] = ld16(Qrow + 48);
    if (mt + 4 < NTILES)
      __builtin_prefetch(Qbase + (size_t)((mt + 4) * 16 + lm) * CQ, 0, 3);
    v8f d = {0.f,0.f,0.f,0.f,0.f,0.f,0.f,0.f};
    d = wmma_bf16(a0, q0, d);
    d = wmma_bf16(a1, q1, d);
    #pragma unroll
    for (int r = 0; r < 8; ++r)                    // one v_fma + one v_exp each
      sm[r] += exp2_raw(fmaf(d[r], LOG2E, -LSE_SHIFT_L2));
  }
  // sum across the 16 lanes sharing each row (xor masks stay within half-wave)
  #pragma unroll
  for (int s = 8; s >= 1; s >>= 1) {
    #pragma unroll
    for (int r = 0; r < 8; ++r)
      sm[r] += __shfl_xor(sm[r], s, 32);
  }
  if (lm == 0) {
    #pragma unroll
    for (int r = 0; r < 8; ++r)
      off2[(size_t)b * NPOS + ntile * 16 + 8 * lh + r] =
          -(LSE_SHIFT_L2 + log2_raw(sm[r]));
  }
}

// ---------------- Kernel 3: O = V * exp2(S*log2e + off2[n]) , fused ----------------
// one wave per 16-col m-tile, owns all 16 c-tiles; streams n in chunks of 32
__global__ void __launch_bounds__(32 * WPB)
attnout_kernel(const u16* __restrict__ QbT, const u16* __restrict__ KbT,
               const u16* __restrict__ Vb, const float* __restrict__ off2,
               float* __restrict__ out) {
  const int lane = threadIdx.x & 31;
  const int wv   = threadIdx.x >> 5;
  const int mtile = blockIdx.x * WPB + wv;
  const int b  = blockIdx.y;
  const int lh = lane >> 4;
  const int lm = lane & 15;

  // B operand: Q tile for this m-tile, fixed
  const u16* Qrow = QbT + ((size_t)b * NPOS + (mtile * 16 + lm)) * CQ + 8 * lh;
  BF16Tile qb0, qb1;
  qb0.q[0] = ld16(Qrow);      qb0.q[1] = ld16(Qrow + 16);
  qb1.q[0] = ld16(Qrow + 32); qb1.q[1] = ld16(Qrow + 48);

  const v8f vz = {0.f,0.f,0.f,0.f,0.f,0.f,0.f,0.f};
  v8f acc[16];
  #pragma unroll
  for (int ci = 0; ci < 16; ++ci) acc[ci] = vz;

  const u16*   Klane = KbT + ((size_t)b * NPOS + lm) * CQ + 8 * lh;
  const float* offb  = off2 + (size_t)b * NPOS + 8 * lh;
  const u16*   Vlane = Vb + ((size_t)b * C_IN + lm) * NPOS + 16 * lh;

  for (int nc = 0; nc < NCHUNKS; ++nc) {
    const int nbase = nc * 32;
    float p0[8], p1[8];
    #pragma unroll
    for (int t = 0; t < 2; ++t) {
      const int n16 = nbase + t * 16;
      const u16* Krow = Klane + (size_t)n16 * CQ;
      BF16Tile a0, a1;
      a0.q[0] = ld16(Krow);      a0.q[1] = ld16(Krow + 16);
      a1.q[0] = ld16(Krow + 32); a1.q[1] = ld16(Krow + 48);
      v8f d = vz;
      d = wmma_bf16(a0, qb0, d);
      d = wmma_bf16(a1, qb1, d);
      float* p = t ? p1 : p0;
      #pragma unroll
      for (int r = 0; r < 8; ++r)               // attn weight = exp(S - logZ[n])
        p[r] = exp2_raw(fmaf(d[r], LOG2E, offb[n16 + r]));
    }
    // Rearrange D-layout (rows r+8*lh across lane halves) -> B-operand layout for P
    BF16Tile pb;
    float ea[8], eb[8];
    #pragma unroll
    for (int r = 0; r < 8; ++r) {
      float x0 = __shfl_xor(p0[r], 16, 32);
      float x1 = __shfl_xor(p1[r], 16, 32);
      ea[r] = lh ? x1 : p0[r];   // element r     (k = 16*lh + r)
      eb[r] = lh ? p1[r] : x0;   // element 8 + r (k = 16*lh + 8 + r)
    }
    #pragma unroll
    for (int i = 0; i < 4; ++i) {
      pb.u[i]     = ((u32)f2bf(ea[2*i+1]) << 16) | f2bf(ea[2*i]);
      pb.u[4 + i] = ((u32)f2bf(eb[2*i+1]) << 16) | f2bf(eb[2*i]);
    }
    // O[c-tiles, m-tile] += V_tile (A) x P (B)
    const u16* Vrow = Vlane + nbase;
    #pragma unroll
    for (int ci = 0; ci < 16; ++ci) {
      BF16Tile va;
      va.q[0] = ld16(Vrow);
      va.q[1] = ld16(Vrow + 8);
      acc[ci] = wmma_bf16(va, pb, acc[ci]);
      Vrow += (size_t)16 * NPOS;
    }
  }
  // epilogue: D layout -> out[b][c][m] f32 (coalesced 16-lane runs)
  float* orow = out + ((size_t)b * C_IN + 8 * lh) * NPOS + mtile * 16 + lm;
  #pragma unroll
  for (int ci = 0; ci < 16; ++ci) {
    #pragma unroll
    for (int r = 0; r < 8; ++r)
      orow[(size_t)(ci * 16 + r) * NPOS] = acc[ci][r];
  }
}

extern "C" void kernel_launch(void* const* d_in, const int* in_sizes, int n_in,
                              void* d_out, int out_size, void* d_ws, size_t ws_size,
                              hipStream_t stream) {
  (void)in_sizes; (void)n_in; (void)out_size; (void)ws_size;
  const float* x  = (const float*)d_in[0];
  const float* W1 = (const float*)d_in[1];
  const float* b1 = (const float*)d_in[2];
  const float* W2 = (const float*)d_in[3];
  const float* b2 = (const float*)d_in[4];
  const float* W3 = (const float*)d_in[5];
  const float* b3 = (const float*)d_in[6];
  float* out = (float*)d_out;

  // workspace carve-up (all 16B aligned): ~9.7 MB total
  u16* QbT = (u16*)d_ws;
  u16* KbT = QbT + (size_t)NB * NPOS * CQ;
  u16* Vb  = KbT + (size_t)NB * NPOS * CQ;
  float* off2 = (float*)(Vb + (size_t)NB * C_IN * NPOS);

  proj_kernel<<<dim3((NPOS + 255) / 256, 2 * CQ + C_IN, NB), 256, 0, stream>>>(
      x, W1, b1, W2, b2, W3, b3, QbT, KbT, Vb);
  rowstats_kernel<<<dim3(NTILES / WPB, NB), 32 * WPB, 0, stream>>>(QbT, KbT, off2);
  attnout_kernel<<<dim3(NTILES / WPB, NB), 32 * WPB, 0, stream>>>(QbT, KbT, Vb, off2, out);
}
